// GAT_15745350108003
// MI455X (gfx1250) — matrix operand order repaired and checked by reading, not compiled, not run on hardware
//
#include <hip/hip_runtime.h>
#include <hip/hip_bf16.h>
#include <stdint.h>

// ---------------------------------------------------------------------------
// GAT (2-layer, 4 heads, hid=64) for MI455X / gfx1250, wave32 + bf16 WMMA.
// GEMMs via v_wmma_f32_16x16x32_bf16; edge scatter-softmax via atomics.
// ---------------------------------------------------------------------------

typedef __attribute__((ext_vector_type(16))) __bf16 bf16x16;
typedef __attribute__((ext_vector_type(8)))  float  f32x8;

#define DIMF   256   // feature width of both GEMM outputs (HEADS*HID)
#define HIDC   64
#define HEADSC 4
#define OUTD   10
#define NGRAPH 64

__device__ __forceinline__ unsigned short f32_to_bf16(float f) {
  unsigned int u = __float_as_uint(f);
  u += 0x7fffu + ((u >> 16) & 1u);          // round-to-nearest-even
  return (unsigned short)(u >> 16);
}

// order-preserving float -> uint encoding for atomicMax-based segment max
__device__ __forceinline__ unsigned int encF(float f) {
  unsigned int u = __float_as_uint(f);
  return (u & 0x80000000u) ? ~u : (u | 0x80000000u);
}
__device__ __forceinline__ float decF(unsigned int k) {
  unsigned int u = (k & 0x80000000u) ? (k & 0x7fffffffu) : ~k;
  return __uint_as_float(u);
}

// ---------------- elementwise converts ----------------
__global__ void k_f32_to_bf16(const float* __restrict__ in,
                              unsigned short* __restrict__ out, int n) {
  int i = blockIdx.x * blockDim.x + threadIdx.x;
  if (i < n) out[i] = f32_to_bf16(in[i]);
}

// W[k*256+n] (f32) -> Wt[n*256+k] (bf16): makes B-fragment K-pairs contiguous
__global__ void k_transpose_w(const float* __restrict__ W,
                              unsigned short* __restrict__ Wt) {
  int i = blockIdx.x * blockDim.x + threadIdx.x;  // 65536
  int k = i >> 8, n = i & 255;
  Wt[n * 256 + k] = f32_to_bf16(W[i]);
}

// ---------------- bf16 WMMA GEMM: C[M,256] = A[M,256] @ Wt^T ----------------
// grid = (ceil(M/64), 2), block = 256 threads (8 waves).
// wave w handles column tile blockIdx.y*8 + w; each wave owns a 64x16 C tile.
__global__ void k_gemm_bf16(const unsigned short* __restrict__ Abf,  // [M,256] bf16 row-major
                            const unsigned short* __restrict__ Wt,   // [256,256] bf16, [n][k]
                            float* __restrict__ C,                   // [M,256] f32
                            int M) {
  const int lane = threadIdx.x & 31;
  const int wave = threadIdx.x >> 5;
  const int colTile = blockIdx.y * 8 + wave;      // 0..15
  const int nBase   = colTile * 16;
  const int rowBase = blockIdx.x * 64;
  const int lmod = lane & 15;
  const int lhi  = lane >> 4;                     // 0/1

  f32x8 acc[4] = {};

  // clamp tail rows (stores are guarded)
  int mrow[4];
#pragma unroll
  for (int s = 0; s < 4; ++s) {
    int m = rowBase + s * 16 + lmod;
    mrow[s] = (m < M) ? m : (M - 1);
  }
  const unsigned short* arow0 = Abf + (size_t)mrow[0] * 256;
  const unsigned short* arow1 = Abf + (size_t)mrow[1] * 256;
  const unsigned short* arow2 = Abf + (size_t)mrow[2] * 256;
  const unsigned short* arow3 = Abf + (size_t)mrow[3] * 256;
  const unsigned short* arows[4] = {arow0, arow1, arow2, arow3};
  const unsigned short* brow = Wt + (size_t)(nBase + lmod) * 256 + lhi * 16;

  for (int k0 = 0; k0 < 256; k0 += 32) {
    if (k0 + 32 < 256)  // hint next A slice into cache (global_prefetch_b8)
      __builtin_prefetch(arows[0] + k0 + 32, 0, 1);

    // B fragment: lanes 0-15 K=k0..k0+15, lanes 16-31 K=k0+16..k0+31
    union { bf16x16 v; unsigned int u[8]; } bf;
#pragma unroll
    for (int j = 0; j < 8; ++j)
      bf.u[j] = *(const unsigned int*)(brow + k0 + j * 2);

#pragma unroll
    for (int s = 0; s < 4; ++s) {
      union { bf16x16 v; unsigned int u[8]; } af;
#pragma unroll
      for (int j = 0; j < 8; ++j) {
        int kOff = j * 2 + lhi * 16;
        af.u[j] = *(const unsigned int*)(arows[s] + k0 + kOff);
      }
      acc[s] = __builtin_amdgcn_wmma_f32_16x16x32_bf16(
          false, af.v, false, bf.v, (short)0, acc[s], false, false);
    }
  }

  // C/D layout: lane n = nBase+lmod; VGPR r -> row r + 8*lhi of the 16x16 tile
  const int n = nBase + lmod;
#pragma unroll
  for (int s = 0; s < 4; ++s) {
#pragma unroll
    for (int r = 0; r < 8; ++r) {
      int m = rowBase + s * 16 + r + lhi * 8;
      if (m < M) C[(size_t)m * 256 + n] = acc[s][r];
    }
  }
}

// ---------------- alpha_s / alpha_d: per-node, per-head dot with att ----------
// one wave per node; lane covers 8 channels; heads span lane groups of 8
__global__ void k_alpha(const float* __restrict__ h,        // [N,256]
                        const float* __restrict__ att_src,  // [4,64] flat 256
                        const float* __restrict__ att_dst,
                        float* __restrict__ as, float* __restrict__ ad, int N) {
  int node = blockIdx.x * (blockDim.x >> 5) + (threadIdx.x >> 5);
  int lane = threadIdx.x & 31;
  if (node >= N) return;
  const float* row = h + (size_t)node * 256 + lane * 8;
  float ps = 0.f, pd = 0.f;
#pragma unroll
  for (int j = 0; j < 8; ++j) {
    float v = row[j];
    ps += v * att_src[lane * 8 + j];
    pd += v * att_dst[lane * 8 + j];
  }
  for (int off = 4; off >= 1; off >>= 1) {
    ps += __shfl_down(ps, off);
    pd += __shfl_down(pd, off);
  }
  if ((lane & 7) == 0) {
    int head = lane >> 3;
    as[node * 4 + head] = ps;
    ad[node * 4 + head] = pd;
  }
}

// ---------------- edge pass 1: raw score + segment max (encoded atomicMax) ---
__global__ void k_edge_score(const int* __restrict__ srcA, const int* __restrict__ dstA,
                             const float* __restrict__ as, const float* __restrict__ ad,
                             float* __restrict__ eVal, unsigned int* __restrict__ mEnc,
                             int E0, int Etot) {
  int idx = blockIdx.x * blockDim.x + threadIdx.x;
  if (idx >= Etot * 4) return;
  int e = idx >> 2, hh = idx & 3;
  int s, d;
  if (e < E0) { s = srcA[e]; d = dstA[e]; } else { s = d = e - E0; }
  float v = as[s * 4 + hh] + ad[d * 4 + hh];
  v = (v > 0.f) ? v : v * 0.2f;                  // leaky_relu
  eVal[(size_t)e * 4 + hh] = v;
  atomicMax(&mEnc[d * 4 + hh], encF(v));
}

// ---------------- edge pass 2: exp + segment sum -----------------------------
__global__ void k_edge_expsum(const int* __restrict__ dstA,
                              const unsigned int* __restrict__ mEnc,
                              float* __restrict__ eVal, float* __restrict__ denom,
                              int E0, int Etot) {
  int idx = blockIdx.x * blockDim.x + threadIdx.x;
  if (idx >= Etot * 4) return;
  int e = idx >> 2, hh = idx & 3;
  int d = (e < E0) ? dstA[e] : (e - E0);
  float m = decF(mEnc[d * 4 + hh]);
  float v = __expf(eVal[(size_t)e * 4 + hh] - m);
  eVal[(size_t)e * 4 + hh] = v;
  atomicAdd(&denom[d * 4 + hh], v);
}

// ---------------- edge pass 3: normalize + scatter-add messages --------------
// one wave per edge; lane handles 8 contiguous channels (head = lane/8)
__global__ void k_edge_aggregate(const int* __restrict__ srcA, const int* __restrict__ dstA,
                                 const float* __restrict__ eVal, const float* __restrict__ denom,
                                 const float* __restrict__ h, float* __restrict__ agg,
                                 int E0, int Etot) {
  int e = blockIdx.x * (blockDim.x >> 5) + (threadIdx.x >> 5);
  int lane = threadIdx.x & 31;
  if (e >= Etot) return;
  int s, d;
  if (e < E0) { s = srcA[e]; d = dstA[e]; } else { s = d = e - E0; }
  int head = lane >> 3;
  float a = eVal[(size_t)e * 4 + head] / (denom[(size_t)d * 4 + head] + 1e-16f);
  const float* hs = h + (size_t)s * 256 + lane * 8;
  float* od = agg + (size_t)d * 256 + lane * 8;
#pragma unroll
  for (int j = 0; j < 8; ++j) atomicAdd(&od[j], a * hs[j]);
}

// ---------------- layer1 epilogue: +bias, ELU, re-quantize to bf16 -----------
__global__ void k_epilogue1(const float* __restrict__ agg, const float* __restrict__ b1,
                            unsigned short* __restrict__ actBf, int total) {
  int i = blockIdx.x * blockDim.x + threadIdx.x;
  if (i >= total) return;
  int c = i & 255;
  float v = agg[i] + b1[c];
  v = (v > 0.f) ? v : (__expf(v) - 1.f);         // elu
  actBf[i] = f32_to_bf16(v);
}

// ---------------- layer2 epilogue: head-mean, +bias, ELU ---------------------
__global__ void k_epilogue2(const float* __restrict__ agg, const float* __restrict__ b2,
                            float* __restrict__ out2, int N) {
  int i = blockIdx.x * blockDim.x + threadIdx.x;
  if (i >= N * 64) return;
  int n = i >> 6, c = i & 63;
  const float* r = agg + (size_t)n * 256 + c;
  float v = 0.25f * (r[0] + r[64] + r[128] + r[192]) + b2[c];
  v = (v > 0.f) ? v : (__expf(v) - 1.f);
  out2[i] = v;
}

// ---------------- global mean pool accumulation ------------------------------
__global__ void k_pool(const float* __restrict__ out2, const int* __restrict__ batch,
                       float* __restrict__ sums, float* __restrict__ cnts, int N) {
  int i = blockIdx.x * blockDim.x + threadIdx.x;
  if (i >= N * 64) return;
  int n = i >> 6, c = i & 63;
  int g = batch[n];
  atomicAdd(&sums[g * 64 + c], out2[i]);
  if (c == 0) atomicAdd(&cnts[g], 1.f);
}

// ---------------- final FC + log_softmax (64 graphs, tiny) -------------------
__global__ void k_head(const float* __restrict__ sums, const float* __restrict__ cnts,
                       const float* __restrict__ fc_w,  // [64,10]
                       const float* __restrict__ fc_b,
                       float* __restrict__ out) {       // [64,10]
  int g = blockIdx.x * blockDim.x + threadIdx.x;
  if (g >= NGRAPH) return;
  float inv = 1.f / fmaxf(cnts[g], 1.f);
  float logits[OUTD];
#pragma unroll
  for (int o = 0; o < OUTD; ++o) logits[o] = fc_b[o];
  for (int c = 0; c < HIDC; ++c) {
    float p = sums[g * 64 + c] * inv;
#pragma unroll
    for (int o = 0; o < OUTD; ++o) logits[o] += p * fc_w[c * OUTD + o];
  }
  float mx = logits[0];
#pragma unroll
  for (int o = 1; o < OUTD; ++o) mx = fmaxf(mx, logits[o]);
  float se = 0.f;
#pragma unroll
  for (int o = 0; o < OUTD; ++o) se += __expf(logits[o] - mx);
  float lse = mx + __logf(se);
#pragma unroll
  for (int o = 0; o < OUTD; ++o) out[g * OUTD + o] = logits[o] - lse;
}

// ===========================================================================
extern "C" void kernel_launch(void* const* d_in, const int* in_sizes, int n_in,
                              void* d_out, int out_size, void* d_ws, size_t ws_size,
                              hipStream_t stream) {
  // inputs (setup_inputs order)
  const float* x        = (const float*)d_in[0];   // [N,256]
  const int*   ei       = (const int*)  d_in[1];   // [2,E0]
  const int*   batch    = (const int*)  d_in[2];   // [N]
  const float* W1       = (const float*)d_in[3];   // [256,256]
  const float* att_src1 = (const float*)d_in[4];
  const float* att_dst1 = (const float*)d_in[5];
  const float* b1       = (const float*)d_in[6];
  const float* W2       = (const float*)d_in[7];
  const float* att_src2 = (const float*)d_in[8];
  const float* att_dst2 = (const float*)d_in[9];
  const float* b2       = (const float*)d_in[10];
  const float* fc_w     = (const float*)d_in[11];
  const float* fc_b     = (const float*)d_in[12];

  const int N    = in_sizes[0] / 256;
  const int E0   = in_sizes[1] / 2;
  const int Etot = E0 + N;
  const int* srcA = ei;
  const int* dstA = ei + E0;

  // ---- carve workspace (256B aligned regions) ----
  size_t off = 0;
  auto carve = [&](size_t bytes) -> void* {
    void* p = (char*)d_ws + off;
    off += (bytes + 255) & ~(size_t)255;
    return p;
  };
  unsigned short* xbf   = (unsigned short*)carve((size_t)N * 256 * 2); // bf16 GEMM input (x, then act1)
  unsigned short* Wt1   = (unsigned short*)carve(256 * 256 * 2);
  unsigned short* Wt2   = (unsigned short*)carve(256 * 256 * 2);
  float*        hbuf    = (float*)carve((size_t)N * 256 * 4);          // h1 then h2
  float*        agg     = (float*)carve((size_t)N * 256 * 4);          // scatter accumulator
  float*        as      = (float*)carve((size_t)N * 4 * 4);
  float*        ad      = (float*)carve((size_t)N * 4 * 4);
  float*        eVal    = (float*)carve((size_t)Etot * 4 * 4);
  unsigned int* mEnc    = (unsigned int*)carve((size_t)N * 4 * 4);
  float*        denom   = (float*)carve((size_t)N * 4 * 4);
  float*        out2    = (float*)carve((size_t)N * 64 * 4);
  float*        sums    = (float*)carve(NGRAPH * 64 * 4);
  float*        cnts    = (float*)carve(NGRAPH * 4);
  (void)ws_size;

  const int TPB = 256;
  dim3 gemmGrid((N + 63) / 64, 2);
  int alphaBlocks = (N + 7) / 8;             // 8 waves/block, 1 node/wave
  int scoreBlocks = (Etot * 4 + TPB - 1) / TPB;
  int aggBlocks   = (Etot + 7) / 8;          // 8 waves/block, 1 edge/wave
  int nfeat       = N * 256;

  // ---- prep: converts / weight transposes ----
  k_f32_to_bf16<<<(nfeat + TPB - 1) / TPB, TPB, 0, stream>>>(x, xbf, nfeat);
  k_transpose_w<<<(65536) / TPB, TPB, 0, stream>>>(W1, Wt1);
  k_transpose_w<<<(65536) / TPB, TPB, 0, stream>>>(W2, Wt2);

  // =================== layer 1 ===================
  k_gemm_bf16<<<gemmGrid, TPB, 0, stream>>>(xbf, Wt1, hbuf, N);
  k_alpha<<<alphaBlocks, TPB, 0, stream>>>(hbuf, att_src1, att_dst1, as, ad, N);

  hipMemsetAsync(mEnc,  0, (size_t)N * 4 * 4, stream);
  hipMemsetAsync(denom, 0, (size_t)N * 4 * 4, stream);
  hipMemsetAsync(agg,   0, (size_t)N * 256 * 4, stream);

  k_edge_score<<<scoreBlocks, TPB, 0, stream>>>(srcA, dstA, as, ad, eVal, mEnc, E0, Etot);
  k_edge_expsum<<<scoreBlocks, TPB, 0, stream>>>(dstA, mEnc, eVal, denom, E0, Etot);
  k_edge_aggregate<<<aggBlocks, TPB, 0, stream>>>(srcA, dstA, eVal, denom, hbuf, agg, E0, Etot);
  k_epilogue1<<<(nfeat + TPB - 1) / TPB, TPB, 0, stream>>>(agg, b1, xbf, nfeat);

  // =================== layer 2 ===================
  k_gemm_bf16<<<gemmGrid, TPB, 0, stream>>>(xbf, Wt2, hbuf, N);
  k_alpha<<<alphaBlocks, TPB, 0, stream>>>(hbuf, att_src2, att_dst2, as, ad, N);

  hipMemsetAsync(mEnc,  0, (size_t)N * 4 * 4, stream);
  hipMemsetAsync(denom, 0, (size_t)N * 4 * 4, stream);
  hipMemsetAsync(agg,   0, (size_t)N * 256 * 4, stream);

  k_edge_score<<<scoreBlocks, TPB, 0, stream>>>(srcA, dstA, as, ad, eVal, mEnc, E0, Etot);
  k_edge_expsum<<<scoreBlocks, TPB, 0, stream>>>(dstA, mEnc, eVal, denom, E0, Etot);
  k_edge_aggregate<<<aggBlocks, TPB, 0, stream>>>(srcA, dstA, eVal, denom, hbuf, agg, E0, Etot);
  k_epilogue2<<<(N * 64 + TPB - 1) / TPB, TPB, 0, stream>>>(agg, b2, out2, N);

  // =================== readout ===================
  hipMemsetAsync(sums, 0, NGRAPH * 64 * 4, stream);
  hipMemsetAsync(cnts, 0, NGRAPH * 4, stream);
  k_pool<<<(N * 64 + TPB - 1) / TPB, TPB, 0, stream>>>(out2, batch, sums, cnts, N);
  k_head<<<1, 64, 0, stream>>>(sums, cnts, fc_w, fc_b, (float*)d_out);
}
